// OnlineConceptCompressionModel_5446018531661
// MI455X (gfx1250) — matrix operand order repaired and checked by reading, not compiled, not run on hardware
//
#include <hip/hip_runtime.h>
#include <hip/hip_bf16.h>
#include <stdint.h>

// ---------------- model constants ----------------
#define H_    1024
#define NH_   16
#define HD_   64
#define S_    1024
#define B_    4
#define CV_   2048
#define FF_   4096
#define TAU_  1.0f
#define BETA_ 0.25f
#define NEGM  -1.0e9f
#define TOK_  (B_ * S_)               // 4096 tokens
#define OUTN  ((size_t)TOK_ * H_)     // 4,194,304 (loss scalar lives after this)

typedef __bf16 bf16;
typedef __attribute__((ext_vector_type(16))) __bf16 v16bf;
typedef __attribute__((ext_vector_type(8)))  __bf16 v8bf;
typedef __attribute__((ext_vector_type(8)))  float  v8f;

// A-fragment (16x32 bf16, ISA 7.12.2): lane m = lane&15, half = lane>>4.
// j=0..7  -> k = half*8 + j ; j=8..15 -> k = 16 + half*8 + (j-8)
// => two contiguous 8-element runs.
// B-fragment (32x16 bf16): lane n = lane&15, half = lane>>4; same k order as A
// => two contiguous 8-element runs.

__device__ __forceinline__ v16bf pack16(const float4 f0, const float4 f1,
                                        const float4 f2, const float4 f3) {
  v16bf r;
  r[0]=(bf16)f0.x; r[1]=(bf16)f0.y; r[2]=(bf16)f0.z; r[3]=(bf16)f0.w;
  r[4]=(bf16)f1.x; r[5]=(bf16)f1.y; r[6]=(bf16)f1.z; r[7]=(bf16)f1.w;
  r[8]=(bf16)f2.x; r[9]=(bf16)f2.y; r[10]=(bf16)f2.z; r[11]=(bf16)f2.w;
  r[12]=(bf16)f3.x; r[13]=(bf16)f3.y; r[14]=(bf16)f3.z; r[15]=(bf16)f3.w;
  return r;
}

// ---------------- LDS-staged bf16 WMMA GEMM, register-blocked ----------------
// C[M,N] = A[M,K](f32->bf16, lda) * W[K,N](f32->bf16, ldb) (+bias[n]) (+res[m,n])
// Block tile 64(M) x 128(N), k-step 32. 8 waves as 2x4; each wave owns a
// 32x32 output (2x2 WMMA tiles -> 4 wmma per k-step, fragment reuse 2x).
// OOB handled by index clamping (garbage only reaches never-stored C tiles).
#define APAD 40   // row stride (bf16 elems): 80B -> 16B aligned, conflict-free
__global__ void __launch_bounds__(256)
gemm_wmma(const float* __restrict__ A, int lda,
          const float* __restrict__ W, int ldb,
          const float* __restrict__ bias,
          const float* __restrict__ res,
          float* __restrict__ C, int ldc,
          int M, int N, int K) {
  __shared__ bf16 als[64][APAD];    // [m][k]
  __shared__ bf16 bls[128][APAD];   // [n][k]  (transposed W tile)

  const int tid  = threadIdx.x;
  const int lane = tid & 31;
  const int wave = tid >> 5;
  const int mblk = blockIdx.y * 64;
  const int nblk = blockIdx.x * 128;
  const int wr   = (wave >> 2) * 32;   // wave M offset in block tile
  const int wc   = (wave & 3) * 32;    // wave N offset in block tile
  const int half = lane >> 4;
  const int lm   = lane & 15;

  // staging assignments
  const int am  = tid >> 2;             // 0..63   (A row)
  const int ak  = (tid & 3) << 3;       // 0,8,16,24 (A k run of 8)
  const int bn  = tid & 127;            // 0..127  (B col)
  const int bk0 = (tid >> 7) << 4;      // 0,16    (B k run of 16)
  const int arow = min(mblk + am, M - 1);
  const int bcol = min(nblk + bn, N - 1);

  v8f acc[4]; acc[0] = {}; acc[1] = {}; acc[2] = {}; acc[3] = {};

  for (int k0 = 0; k0 < K; k0 += 32) {
    // L2 prefetch of next k-step tiles (global_prefetch_b8)
    if (k0 + 32 < K) {
      __builtin_prefetch(A + (size_t)arow * lda + k0 + 32 + ak, 0, 1);
      __builtin_prefetch(W + (size_t)(k0 + 32 + bk0) * ldb + bcol, 0, 1);
    }
    // stage A tile: two coalesced float4 per thread, convert to bf16
    {
      const float* ap = A + (size_t)arow * lda + k0 + ak;
      const float4 v0 = *(const float4*)(ap);
      const float4 v1 = *(const float4*)(ap + 4);
      bf16* dst = &als[am][ak];
      dst[0]=(bf16)v0.x; dst[1]=(bf16)v0.y; dst[2]=(bf16)v0.z; dst[3]=(bf16)v0.w;
      dst[4]=(bf16)v1.x; dst[5]=(bf16)v1.y; dst[6]=(bf16)v1.z; dst[7]=(bf16)v1.w;
    }
    // stage B tile transposed: each dword load is 128-lane contiguous in N
    {
      float tmp[16];
#pragma unroll
      for (int i = 0; i < 16; ++i)
        tmp[i] = W[(size_t)(k0 + bk0 + i) * ldb + bcol];
      bf16* dst = &bls[bn][bk0];
#pragma unroll
      for (int i = 0; i < 16; ++i) dst[i] = (bf16)tmp[i];
    }
    __syncthreads();
    // fragments (contiguous 16B/32B LDS reads), 2 per operand, 4 wmma
    v16bf afrag[2], bfrag[2];
#pragma unroll
    for (int i = 0; i < 2; ++i) {
      const int mrow = wr + i * 16 + lm;
      const v8bf lo = *(const v8bf*)&als[mrow][half * 8];
      const v8bf hi = *(const v8bf*)&als[mrow][16 + half * 8];
      afrag[i] = __builtin_shufflevector(lo, hi,
          0,1,2,3,4,5,6,7,8,9,10,11,12,13,14,15);
      const int ncol = wc + i * 16 + lm;
      const v8bf blo = *(const v8bf*)&bls[ncol][half * 8];
      const v8bf bhi = *(const v8bf*)&bls[ncol][16 + half * 8];
      bfrag[i] = __builtin_shufflevector(blo, bhi,
          0,1,2,3,4,5,6,7,8,9,10,11,12,13,14,15);
    }
#pragma unroll
    for (int ti = 0; ti < 2; ++ti)
#pragma unroll
      for (int tj = 0; tj < 2; ++tj)
        acc[ti * 2 + tj] = __builtin_amdgcn_wmma_f32_16x16x32_bf16(
            false, afrag[ti], false, bfrag[tj], (short)0, acc[ti * 2 + tj],
            false, false);
    __syncthreads();
  }

  const bool full = (mblk + 64 <= M) && (nblk + 128 <= N);
#pragma unroll
  for (int ti = 0; ti < 2; ++ti)
#pragma unroll
    for (int tj = 0; tj < 2; ++tj) {
      const v8f a = acc[ti * 2 + tj];
#pragma unroll
      for (int r = 0; r < 8; ++r) {
        const int m = mblk + wr + ti * 16 + half * 8 + r;
        const int n = nblk + wc + tj * 16 + lm;
        if (full) {
          float v = a[r];
          if (bias) v += bias[n];
          if (res)  v += res[(size_t)m * ldc + n];
          C[(size_t)m * ldc + n] = v;
        } else if (m < M && n < N) {
          float v = a[r];
          if (bias) v += bias[n];
          if (res)  v += res[(size_t)m * ldc + n];
          C[(size_t)m * ldc + n] = v;
        }
      }
    }
}

// ---------------- flash attention (one wave per (qtile, head, batch)) ----------------
// V blocks are staged LDS-side via CDNA5 async copies (global_load_async_to_lds_b128,
// tracked by ASYNCcnt) issued at the top of each key-block iteration; the copy
// overlaps the K loads, the two score WMMAs and the softmax VALU work, and is
// only waited on (s_wait_asynccnt 0) right before the P x V WMMAs.
#define VPAD 68   // f32 row stride: 272B (16B-aligned rows for async b128)
__global__ void __launch_bounds__(32)
attn_kernel(const float* __restrict__ q, const float* __restrict__ k,
            const float* __restrict__ v, const int* __restrict__ valid,
            float* __restrict__ out) {
  __shared__ float sc[16][33];
  __shared__ bf16  pb[16][APAD];
  __shared__ float fac[16];
  __shared__ float lrow[16];
  __shared__ float vls[32][VPAD];   // async-staged V block [key][dim]

  const int lane = threadIdx.x & 31;
  const int half = lane >> 4;
  const int lm   = lane & 15;
  const int qt = blockIdx.x, hh = blockIdx.y, b = blockIdx.z;
  const int q0 = qt * 16;
  const float scale = 0.125f; // 1/sqrt(HD)

  // Q fragments (two K-chunks of 32 over HD=64), pre-scaled; runs are contiguous
  v16bf qa0, qa1;
  {
    const float* qp = q + ((size_t)(b * S_ + q0 + lm) * NH_ + hh) * HD_;
    float4 f[8];
    f[0] = *(const float4*)(qp + half * 8);
    f[1] = *(const float4*)(qp + half * 8 + 4);
    f[2] = *(const float4*)(qp + 16 + half * 8);
    f[3] = *(const float4*)(qp + 16 + half * 8 + 4);
    f[4] = *(const float4*)(qp + 32 + half * 8);
    f[5] = *(const float4*)(qp + 32 + half * 8 + 4);
    f[6] = *(const float4*)(qp + 48 + half * 8);
    f[7] = *(const float4*)(qp + 48 + half * 8 + 4);
#pragma unroll
    for (int i = 0; i < 8; ++i) {
      f[i].x *= scale; f[i].y *= scale; f[i].z *= scale; f[i].w *= scale;
    }
    qa0 = pack16(f[0], f[1], f[2], f[3]);
    qa1 = pack16(f[4], f[5], f[6], f[7]);
  }

  float mi = -3.0e38f, li = 0.0f;
  v8f acc[4]; acc[0] = {}; acc[1] = {}; acc[2] = {}; acc[3] = {};

  for (int kb = 0; kb < S_ / 32; ++kb) {
    // kick off async V-block copy (each lane owns one key row: 16 x b128)
    {
      const int key = kb * 32 + lane;
      const float* vp = v + ((size_t)(b * S_ + key) * NH_ + hh) * HD_;
      const unsigned laddr = (unsigned)(uintptr_t)&vls[lane][0];
#pragma unroll
      for (int i = 0; i < 16; ++i) {
        asm volatile("global_load_async_to_lds_b128 %0, %1, off"
                     :: "v"(laddr + i * 16), "v"(vp + i * 4) : "memory");
      }
    }
    // scores for 32 keys (two 16-col tiles)
#pragma unroll
    for (int ct = 0; ct < 2; ++ct) {
      const int key = kb * 32 + ct * 16 + lm;
      const float* kp = k + ((size_t)(b * S_ + key) * NH_ + hh) * HD_ + half * 8;
      // B-fragment k order matches A: two 8-float runs at half*8 and 16+half*8
      const v16bf kf0 = pack16(*(const float4*)(kp),
                               *(const float4*)(kp + 4),
                               *(const float4*)(kp + 16),
                               *(const float4*)(kp + 20));
      const v16bf kf1 = pack16(*(const float4*)(kp + 32),
                               *(const float4*)(kp + 36),
                               *(const float4*)(kp + 48),
                               *(const float4*)(kp + 52));
      v8f s = {};
      s = __builtin_amdgcn_wmma_f32_16x16x32_bf16(false, qa0, false, kf0, (short)0, s, false, false);
      s = __builtin_amdgcn_wmma_f32_16x16x32_bf16(false, qa1, false, kf1, (short)0, s, false, false);
#pragma unroll
      for (int r = 0; r < 8; ++r) {
        const int qrow = q0 + half * 8 + r;
        const bool ok = (key <= qrow) && (valid[b * S_ + key] != 0) && (valid[b * S_ + qrow] != 0);
        sc[half * 8 + r][ct * 16 + lm] = s[r] + (ok ? 0.0f : NEGM);
      }
    }
    __syncthreads();
    if (lane < 16) {
      const int row = lane;
      float tm = -3.0e38f;
      for (int c = 0; c < 32; ++c) tm = fmaxf(tm, sc[row][c]);
      const float newm = fmaxf(mi, tm);
      const float f = __expf(mi - newm);
      float s = 0.0f;
      for (int c = 0; c < 32; ++c) {
        const float p = __expf(sc[row][c] - newm);
        s += p;
        pb[row][c] = (bf16)p;
      }
      li = li * f + s;
      mi = newm;
      fac[row] = f;
    }
    __syncthreads();
    // P fragment from LDS (A layout = two contiguous 8-elem runs)
    const v8bf plo = *(const v8bf*)&pb[lm][half * 8];
    const v8bf phi = *(const v8bf*)&pb[lm][16 + half * 8];
    const v16bf pa = __builtin_shufflevector(plo, phi,
        0,1,2,3,4,5,6,7,8,9,10,11,12,13,14,15);
    // V block must have landed in LDS by now
    asm volatile("s_wait_asynccnt 0x0" ::: "memory");
#pragma unroll
    for (int t = 0; t < 4; ++t) {
      v16bf vb;
#pragma unroll
      for (int j = 0; j < 16; ++j)
        vb[j] = (bf16)vls[(j < 8 ? half * 8 + j : 8 + half * 8 + j)][t * 16 + lm];
#pragma unroll
      for (int r = 0; r < 8; ++r) acc[t][r] *= fac[half * 8 + r];
      acc[t] = __builtin_amdgcn_wmma_f32_16x16x32_bf16(false, pa, false, vb, (short)0, acc[t], false, false);
    }
    __syncthreads();
  }
  if (lane < 16) lrow[lane] = li;
  __syncthreads();
#pragma unroll
  for (int t = 0; t < 4; ++t)
#pragma unroll
    for (int r = 0; r < 8; ++r) {
      const int m = half * 8 + r;
      out[((size_t)(b * S_ + q0 + m) * NH_ + hh) * HD_ + t * 16 + lm] = acc[t][r] / lrow[m];
    }
}

// ---------------- elementwise / reduction kernels ----------------
__global__ void embed_kernel(const int* __restrict__ ids, const float* __restrict__ emb,
                             float* __restrict__ x) {
  const int t = blockIdx.x;
  const int id = ids[t];
  for (int e = threadIdx.x; e < H_; e += 256)
    x[(size_t)t * H_ + e] = emb[(size_t)id * H_ + e];
}

__global__ void rmsnorm_kernel(const float* __restrict__ x, const float* __restrict__ w,
                               float* __restrict__ y) {
  __shared__ float red[256];
  const int row = blockIdx.x;
  float s = 0.0f;
  for (int i = threadIdx.x; i < H_; i += 256) {
    const float t = x[(size_t)row * H_ + i];
    s += t * t;
  }
  red[threadIdx.x] = s; __syncthreads();
  for (int off = 128; off > 0; off >>= 1) {
    if (threadIdx.x < off) red[threadIdx.x] += red[threadIdx.x + off];
    __syncthreads();
  }
  const float r = rsqrtf(red[0] / (float)H_ + 1e-6f);
  for (int i = threadIdx.x; i < H_; i += 256)
    y[(size_t)row * H_ + i] = x[(size_t)row * H_ + i] * r * w[i];
}

__global__ void rope_kernel(float* __restrict__ x, const int* __restrict__ pos) {
  const int t = blockIdx.x;           // token
  const int hh = threadIdx.x >> 5;    // head
  const int i  = threadIdx.x & 31;    // rotation pair index
  const int p  = pos ? pos[t] : (t & (S_ - 1));
  const float inv = __expf(-9.210340371976184f * (2.0f * i) / (float)HD_); // 10000^(-2i/HD)
  const float ang = (float)p * inv;
  const float c = __cosf(ang), s = __sinf(ang);
  const size_t base = ((size_t)t * NH_ + hh) * HD_;
  const float x1 = x[base + i], x2 = x[base + 32 + i];
  x[base + i]      = x1 * c - x2 * s;
  x[base + 32 + i] = x2 * c + x1 * s;
}

__global__ void silu_mul_kernel(float* __restrict__ g, const float* __restrict__ u, size_t n) {
  size_t i = (size_t)blockIdx.x * blockDim.x + threadIdx.x;
  const size_t stride = (size_t)gridDim.x * blockDim.x;
  for (; i < n; i += stride) {
    const float x = g[i];
    g[i] = (x / (1.0f + __expf(-x))) * u[i];
  }
}

// gumbel softmax over 2049 logits per token (in place) + argmax
__global__ void gumbel_softmax_kernel(float* __restrict__ z, const float* __restrict__ gum,
                                      int* __restrict__ chosen) {
  __shared__ float smax[256]; __shared__ int sarg[256]; __shared__ float ssum[256];
  const int t = blockIdx.x, tid = threadIdx.x;
  const int NV = CV_ + 1;
  const float invtau = 1.0f / TAU_;
  float mx = -3.0e38f; int arg = 0;
  for (int j = tid; j < NV; j += 256) {
    const float tv = (z[(size_t)t * NV + j] + gum[(size_t)t * NV + j]) * invtau;
    if (tv > mx) { mx = tv; arg = j; }
  }
  smax[tid] = mx; sarg[tid] = arg; __syncthreads();
  for (int off = 128; off > 0; off >>= 1) {
    if (tid < off) {
      const float o = smax[tid + off]; const int oa = sarg[tid + off];
      if (o > smax[tid] || (o == smax[tid] && oa < sarg[tid])) { smax[tid] = o; sarg[tid] = oa; }
    }
    __syncthreads();
  }
  const float gmax = smax[0]; const int garg = sarg[0];
  float s = 0.0f;
  for (int j = tid; j < NV; j += 256) {
    const float tv = (z[(size_t)t * NV + j] + gum[(size_t)t * NV + j]) * invtau;
    const float e = __expf(tv - gmax);
    z[(size_t)t * NV + j] = e;
    s += e;
  }
  ssum[tid] = s; __syncthreads();
  for (int off = 128; off > 0; off >>= 1) {
    if (tid < off) ssum[tid] += ssum[tid + off];
    __syncthreads();
  }
  const float inv = 1.0f / ssum[0];
  for (int j = tid; j < NV; j += 256) z[(size_t)t * NV + j] *= inv;
  if (tid == 0) chosen[t] = garg;
}

// e_hard = cemb[min(chosen,CV-1)] * is_comp ; loss += (e_soft - e_hard)^2
__global__ void ehard_loss_kernel(const int* __restrict__ chosen, const int* __restrict__ valid,
                                  const float* __restrict__ es, const float* __restrict__ cemb,
                                  float* __restrict__ eh, float* __restrict__ loss) {
  __shared__ float red[256];
  const int t = blockIdx.x;
  const int c = chosen[t];
  const int comp = (c != CV_) && (valid[t] != 0);
  const int cc = c < CV_ ? c : CV_ - 1;
  float local = 0.0f;
  for (int e = threadIdx.x; e < H_; e += 256) {
    const float hv = comp ? cemb[(size_t)cc * H_ + e] : 0.0f;
    eh[(size_t)t * H_ + e] = hv;
    const float d = es[(size_t)t * H_ + e] - hv;
    local += d * d;
  }
  red[threadIdx.x] = local; __syncthreads();
  for (int off = 128; off > 0; off >>= 1) {
    if (threadIdx.x < off) red[threadIdx.x] += red[threadIdx.x + off];
    __syncthreads();
  }
  if (threadIdx.x == 0) atomicAdd(loss, red[0]);
}

// stable pack per batch: comp positions first (Hillis-Steele scan in LDS)
__global__ void __launch_bounds__(1024)
pack_kernel(const int* __restrict__ chosen, const int* __restrict__ valid,
            int* __restrict__ order, int* __restrict__ cpos,
            int* __restrict__ cvalid, int* __restrict__ counts) {
  __shared__ int pc[S_];
  const int b = blockIdx.x, i = threadIdx.x;
  const int comp = (chosen[b * S_ + i] != CV_) && (valid[b * S_ + i] != 0);
  pc[i] = comp; __syncthreads();
  for (int off = 1; off < S_; off <<= 1) {
    const int vv = (i >= off) ? pc[i - off] : 0;
    __syncthreads();
    pc[i] += vv;
    __syncthreads();
  }
  const int total = pc[S_ - 1];
  const int excl = pc[i] - comp;
  const int dest = comp ? excl : (total + (i - excl));
  order[b * S_ + dest] = i;
  __syncthreads();
  const int cv = (i < total) ? 1 : 0;
  cvalid[b * S_ + i] = cv;
  cpos[b * S_ + i] = cv ? order[b * S_ + i] : 0;
  if (i == 0) counts[b] = total;
}

__global__ void gather_kernel(const float* __restrict__ eh, const int* __restrict__ order,
                              const int* __restrict__ counts, float* __restrict__ cin) {
  const int t = blockIdx.x;
  const int b = t / S_, j = t % S_;
  const int src = order[b * S_ + j];
  const int ok = (j < counts[b]);
  for (int e = threadIdx.x; e < H_; e += 256)
    cin[(size_t)t * H_ + e] = ok ? eh[(size_t)(b * S_ + src) * H_ + e] : 0.0f;
}

__global__ void final_kernel(const float* __restrict__ hmid, const int* __restrict__ cvalid,
                             float* __restrict__ out) {
  const int t = blockIdx.x;
  const float m = cvalid[t] ? 1.0f : 0.0f;
  for (int e = threadIdx.x; e < H_; e += 256)
    out[(size_t)t * H_ + e] = hmid[(size_t)t * H_ + e] * m;
}

__global__ void scalar_zero_kernel(float* p) { *p = 0.0f; }
__global__ void loss_finalize_kernel(float* p) {
  *p *= (1.0f + BETA_) / (float)((size_t)TOK_ * H_);
}

// ---------------- host orchestration ----------------
extern "C" void kernel_launch(void* const* d_in, const int* in_sizes, int n_in,
                              void* d_out, int out_size, void* d_ws, size_t ws_size,
                              hipStream_t stream) {
  (void)in_sizes; (void)n_in; (void)out_size; (void)ws_size;
  const int*   input_ids = (const int*)d_in[0];
  const int*   amask     = (const int*)d_in[1];
  const float* embed_w   = (const float*)d_in[2];
  const float* ln1       = (const float*)d_in[3];
  const float* wq        = (const float*)d_in[4];
  const float* wk        = (const float*)d_in[5];
  const float* wv        = (const float*)d_in[6];
  const float* wo        = (const float*)d_in[7];
  const float* ln2       = (const float*)d_in[8];
  const float* wg        = (const float*)d_in[9];
  const float* wu        = (const float*)d_in[10];
  const float* wd        = (const float*)d_in[11];
  const float* head_w    = (const float*)d_in[12];
  const float* head_b    = (const float*)d_in[13];
  const float* cemb      = (const float*)d_in[14];
  const float* gum       = (const float*)d_in[15];
  float* out = (float*)d_out;
  float* ws  = (float*)d_ws;

  const size_t M4 = (size_t)TOK_ * H_;      // 4M floats
  float* hbuf = ws + 0 * M4;
  float* hn   = ws + 1 * M4;
  float* qb   = ws + 2 * M4;
  float* kbuf = ws + 3 * M4;
  float* vb   = ws + 4 * M4;
  float* ao   = ws + 5 * M4;
  float* gbuf = ws + 6 * M4;                // 4*M4 (FF activations)
  float* ubuf = ws + 10 * M4;               // 4*M4
  float* zbuf = ws + 6 * M4;                // overlays g (disjoint lifetime)
  float* es   = ws + 9 * M4;                // overlays g tail
  float* eh   = ws + 10 * M4;               // overlays u (disjoint lifetime)
  int*   ip   = (int*)(ws + 14 * M4);
  int* chosen = ip;
  int* order  = ip + TOK_;
  int* cpos   = ip + 2 * TOK_;
  int* cvalid = ip + 3 * TOK_;
  int* counts = ip + 4 * TOK_;

  auto gemm = [&](const float* A, int lda, const float* W, int ldb,
                  const float* bias, const float* res, float* C, int ldc,
                  int M, int N, int K) {
    dim3 grid((N + 127) / 128, (M + 63) / 64);
    gemm_wmma<<<grid, 256, 0, stream>>>(A, lda, W, ldb, bias, res, C, ldc, M, N, K);
  };

  auto run_block = [&](int layer, const int* pos, const int* valid) {
    rmsnorm_kernel<<<TOK_, 256, 0, stream>>>(hbuf, ln1 + (size_t)layer * H_, hn);
    gemm(hn, H_, wq + (size_t)layer * H_ * H_, H_, nullptr, nullptr, qb,   H_, TOK_, H_, H_);
    gemm(hn, H_, wk + (size_t)layer * H_ * H_, H_, nullptr, nullptr, kbuf, H_, TOK_, H_, H_);
    gemm(hn, H_, wv + (size_t)layer * H_ * H_, H_, nullptr, nullptr, vb,   H_, TOK_, H_, H_);
    rope_kernel<<<TOK_, NH_ * 32, 0, stream>>>(qb, pos);
    rope_kernel<<<TOK_, NH_ * 32, 0, stream>>>(kbuf, pos);
    attn_kernel<<<dim3(S_ / 16, NH_, B_), 32, 0, stream>>>(qb, kbuf, vb, valid, ao);
    gemm(ao, H_, wo + (size_t)layer * H_ * H_, H_, nullptr, hbuf, hbuf, H_, TOK_, H_, H_);
    rmsnorm_kernel<<<TOK_, 256, 0, stream>>>(hbuf, ln2 + (size_t)layer * H_, hn);
    gemm(hn, H_, wg + (size_t)layer * H_ * FF_, FF_, nullptr, nullptr, gbuf, FF_, TOK_, FF_, H_);
    gemm(hn, H_, wu + (size_t)layer * H_ * FF_, FF_, nullptr, nullptr, ubuf, FF_, TOK_, FF_, H_);
    silu_mul_kernel<<<4096, 256, 0, stream>>>(gbuf, ubuf, (size_t)TOK_ * FF_);
    gemm(gbuf, FF_, wd + (size_t)layer * FF_ * H_, H_, nullptr, hbuf, hbuf, H_, TOK_, H_, FF_);
  };

  // embed
  embed_kernel<<<TOK_, 256, 0, stream>>>(input_ids, embed_w, hbuf);
  // shallow blocks (pos = arange, mask = causal & attention_mask)
  run_block(0, nullptr, amask);
  run_block(1, nullptr, amask);

  // concept head: logits -> gumbel softmax -> chosen / z
  gemm(hbuf, H_, head_w, CV_ + 1, head_b, nullptr, zbuf, CV_ + 1, TOK_, CV_ + 1, H_);
  gumbel_softmax_kernel<<<TOK_, 256, 0, stream>>>(zbuf, gum, chosen);
  // e_soft = z[:, :CV] @ concept_emb
  gemm(zbuf, CV_ + 1, cemb, H_, nullptr, nullptr, es, H_, TOK_, H_, CV_);
  // e_hard + VQ loss (cea == e_hard numerically; straight-through cancels in fwd)
  scalar_zero_kernel<<<1, 1, 0, stream>>>(out + OUTN);
  ehard_loss_kernel<<<TOK_, 256, 0, stream>>>(chosen, amask, es, cemb, eh, out + OUTN);
  // stable pack to front + gather compressed inputs into hbuf
  pack_kernel<<<B_, S_, 0, stream>>>(chosen, amask, order, cpos, cvalid, counts);
  gather_kernel<<<TOK_, 256, 0, stream>>>(eh, order, counts, hbuf);

  // mid blocks on compressed stream (pos = cpos, mask from cvalid)
  run_block(2, cpos, cvalid);
  run_block(3, cpos, cvalid);

  // outputs: cmid * cvalid, then scale loss
  final_kernel<<<TOK_, 256, 0, stream>>>(hbuf, cvalid, out);
  loss_finalize_kernel<<<1, 1, 0, stream>>>(out + OUTN);
}
